// ChunkedAttention_65687229825497
// MI455X (gfx1250) — matrix operand order repaired and checked by reading, not compiled, not run on hardware
//
#include <hip/hip_runtime.h>

#define DM   1024
#define NH   16
#define DH   64
#define TT   4096
#define BBATCH 4
#define CH   128
#define MTOT (BBATCH * TT)   // 16384 rows

typedef __attribute__((ext_vector_type(16))) __bf16 bf16x16;
typedef __attribute__((ext_vector_type(4)))  __bf16 bf16x4;
typedef __attribute__((ext_vector_type(8)))  float  f32x8;
typedef __attribute__((ext_vector_type(4)))  unsigned u32x4;
typedef __attribute__((ext_vector_type(8)))  unsigned u32x8;

union FragU { bf16x16 v; uint4 q[2]; };

// A-matrix fragment 16x32 bf16 (ISA 7.12.2): lane = row M (mod 16);
// lanes 0-15 hold K=0..7 (q[0]) and K=16..23 (q[1]); lanes 16-31 hold K=8..15 / K=24..31.
__device__ __forceinline__ bf16x16 frag_a(const __bf16* base, int stride, int lane) {
  const int r = lane & 15, h = lane >> 4;
  FragU f;
  f.q[0] = *(const uint4*)(base + r * stride + h * 8);
  f.q[1] = *(const uint4*)(base + r * stride + h * 8 + 16);
  return f.v;
}

// B-matrix fragment 32x16 bf16 (KxN): lane = col N; lanes 0-15 hold K=0..7 / K=16..23, lanes 16-31 K=8..15 / K=24..31.
// Source rows hold a column of B contiguously in K (NT layout), so loads are two b128s.
__device__ __forceinline__ bf16x16 frag_b(const __bf16* base, int stride, int lane) {
  const int n = lane & 15, h = lane >> 4;
  FragU f;
  f.q[0] = *(const uint4*)(base + n * stride + h * 8);
  f.q[1] = *(const uint4*)(base + n * stride + h * 8 + 16);
  return f.v;
}

__device__ __forceinline__ f32x8 wmma_bf16(bf16x16 a, bf16x16 b, f32x8 c) {
  return __builtin_amdgcn_wmma_f32_16x16x32_bf16(false, a, false, b, (short)0, c,
                                                 false, false);
}

// ---------------------------------------------------------------------------
// GEMM: Y[r,c] = sum_k A[r,k] * W[c,k]   (A: MTOTxDM, W: DMxDM row-major, NT)
// Block tile 128x128, BK=64, 8 waves (2 M x 4 N), each wave 64x32 C-tile.
// Next K-slab is prefetched into registers while the current slab computes.
// SPLITHEAD: write bf16 into (B, NH, T, DH) layout; else fp32 row-major (M, DM).
// ---------------------------------------------------------------------------
constexpr int LDG = 72;                   // 64 + 8 bf16 pad; 144B rows (16B aligned)

__device__ __forceinline__ void ld_f32(const float* P, int rBase, int k0, int tid, float4* r) {
  for (int i = 0; i < 8; ++i) {
    int id = tid + i * 256;               // 2048 x float4 (128 x 64 fp32 tile)
    r[i] = *(const float4*)(P + (size_t)(rBase + (id >> 4)) * DM + k0 + (id & 15) * 4);
  }
}
__device__ __forceinline__ void st_f32(__bf16* S, int tid, const float4* r) {
  for (int i = 0; i < 8; ++i) {
    int id = tid + i * 256;
    bf16x4 h;                             // hardware RNE converts (v_cvt_pk_bf16_f32)
    h.x = (__bf16)r[i].x; h.y = (__bf16)r[i].y;
    h.z = (__bf16)r[i].z; h.w = (__bf16)r[i].w;
    *(bf16x4*)(S + (id >> 4) * LDG + (id & 15) * 4) = h;
  }
}
__device__ __forceinline__ void ld_bf16(const __bf16* P, int rBase, int k0, int tid, uint4* r) {
  for (int i = 0; i < 4; ++i) {
    int id = tid + i * 256;               // 1024 x b128 (128 x 64 bf16 tile)
    r[i] = *(const uint4*)(P + (size_t)(rBase + (id >> 3)) * DM + k0 + (id & 7) * 8);
  }
}
__device__ __forceinline__ void st_bf16(__bf16* S, int tid, const uint4* r) {
  for (int i = 0; i < 4; ++i) {
    int id = tid + i * 256;
    *(uint4*)(S + (id >> 3) * LDG + (id & 7) * 8) = r[i];
  }
}

template<bool A_BF16, bool SPLITHEAD>
__global__ __launch_bounds__(256)
void gemm_bf16(const void* __restrict__ Ap, const float* __restrict__ Wp,
               void* __restrict__ Op) {
  __shared__ __bf16 As[128 * LDG];        // 18 KB
  __shared__ __bf16 Bs[128 * LDG];        // 18 KB

  const int tid = threadIdx.x, lane = tid & 31, wave = tid >> 5;
  const int rowBase = blockIdx.y * 128, colBase = blockIdx.x * 128;
  const int wm = (wave >> 2) * 64, wn = (wave & 3) * 32;

  f32x8 acc[4][2] = {};
  float4 raf[8]; uint4 rab[4]; float4 rbf[8];

  // prologue: stage k-slab 0
  if (A_BF16) { ld_bf16((const __bf16*)Ap, rowBase, 0, tid, rab); }
  else        { ld_f32((const float*)Ap, rowBase, 0, tid, raf); }
  ld_f32(Wp, colBase, 0, tid, rbf);
  if (A_BF16) st_bf16(As, tid, rab); else st_f32(As, tid, raf);
  st_f32(Bs, tid, rbf);
  __syncthreads();

  for (int k0 = 0; k0 < DM; k0 += 64) {
    const bool nxt = (k0 + 64 < DM);
    if (nxt) {                            // global loads for slab i+1 in flight
      if (A_BF16) { ld_bf16((const __bf16*)Ap, rowBase, k0 + 64, tid, rab); }
      else        { ld_f32((const float*)Ap, rowBase, k0 + 64, tid, raf); }
      ld_f32(Wp, colBase, k0 + 64, tid, rbf);
    }
    for (int kk = 0; kk < 64; kk += 32) { // compute slab i
      bf16x16 af[4], bfr[2];
      for (int mt = 0; mt < 4; ++mt) af[mt]  = frag_a(As + (wm + mt * 16) * LDG + kk, LDG, lane);
      for (int nt = 0; nt < 2; ++nt) bfr[nt] = frag_b(Bs + (wn + nt * 16) * LDG + kk, LDG, lane);
      for (int mt = 0; mt < 4; ++mt)
        for (int nt = 0; nt < 2; ++nt)
          acc[mt][nt] = wmma_bf16(af[mt], bfr[nt], acc[mt][nt]);
    }
    if (nxt) {
      __syncthreads();
      if (A_BF16) st_bf16(As, tid, rab); else st_f32(As, tid, raf);
      st_f32(Bs, tid, rbf);
      __syncthreads();
    }
  }

  // C/D layout: vgpr i holds row i (lanes 0-15) / row i+8 (lanes 16-31), col = lane&15
  const int half = lane >> 4, ln = lane & 15;
  for (int mt = 0; mt < 4; ++mt)
    for (int nt = 0; nt < 2; ++nt)
      for (int i = 0; i < 8; ++i) {
        int row = rowBase + wm + mt * 16 + i + 8 * half;
        int col = colBase + wn + nt * 16 + ln;
        float v = acc[mt][nt][i];
        if (SPLITHEAD) {
          int b = row >> 12, t = row & (TT - 1);
          int h = col >> 6,  d = col & 63;
          ((__bf16*)Op)[((((size_t)b * NH + h) * TT + t) << 6) + d] = (__bf16)v;
        } else {
          ((float*)Op)[((size_t)row << 10) + col] = v;
        }
      }
}

// ---------------------------------------------------------------------------
// Flash attention, causal, chunk = 128 queries per block, 8 waves x 16 q-rows.
// K tile is DMA'd by the Tensor Data Mover (tensor_load_to_lds) with LDS row
// padding 64->72 done by the TDM pad fields; V is loaded by VMEM and stored
// transposed. P overlays the K region once S is consumed. LDS total 52224 B.
// ---------------------------------------------------------------------------
__global__ __launch_bounds__(256)
void flash_kernel(const unsigned short* __restrict__ Qg,
                  const unsigned short* __restrict__ Kg,
                  const unsigned short* __restrict__ Vg,
                  unsigned short* __restrict__ Og) {
  constexpr int LDK = 72, LDV = 136, LDP = 136;
  constexpr int PS_BYTES = 8 * 16 * LDP * 2;          // 34816
  __shared__ __align__(16) char smem[PS_BYTES + 64 * LDV * 2];
  __bf16* Ks = (__bf16*)smem;                          // phase 1
  __bf16* Ps = (__bf16*)smem;                          // phase 2 (overlay)
  __bf16* Vt = (__bf16*)(smem + PS_BYTES);             // V transposed: [d][t]

  const int tid = threadIdx.x, lane = tid & 31, wave = tid >> 5;
  const int half = lane >> 4, ln = lane & 15;
  const int qc = blockIdx.x, bh = blockIdx.y;

  const __bf16* Qh = (const __bf16*)Qg + (size_t)bh * TT * DH;
  const __bf16* Kh = (const __bf16*)Kg + (size_t)bh * TT * DH;
  const __bf16* Vh = (const __bf16*)Vg + (size_t)bh * TT * DH;

  const int qrow = qc * CH + wave * 16;
  bf16x16 qf0 = frag_a(Qh + (size_t)qrow * DH,      DH, lane);   // k = 0..31
  bf16x16 qf1 = frag_a(Qh + (size_t)qrow * DH + 32, DH, lane);   // k = 32..63

  float m[8], l[8];
  f32x8 o[4] = {};
  for (int i = 0; i < 8; ++i) { m[i] = -__builtin_inff(); l[i] = 0.f; }

  for (int kc = 0; kc <= qc; ++kc) {
    __syncthreads();        // previous iteration's readers done with Ks/Ps/Vt

    // --- K tile (128 x 64 bf16) via Tensor Data Mover, one descriptor/wave0.
    // D# group0: count=1 | lds_addr | global_addr | type=2.
    // D# group1: data_size=2B, pad_enable, pad_interval=32 dwords (one 64-elem
    // row), pad_amount=4 dwords (8 bf16) -> LDS row stride 72 = LDK.
    // tile 64 x 128, tensor_dim0=64, dim0_stride=64 (contiguous rows).
    if (wave == 0) {
      unsigned long long ga =
          (unsigned long long)(uintptr_t)(const void*)(Kh + (size_t)kc * CH * DH);
      unsigned lds = (unsigned)(unsigned long long)(uintptr_t)(void*)Ks; // low 32 = LDS offset
      u32x4 g0 = { 1u, lds, (unsigned)ga,
                   (unsigned)((ga >> 32) & 0x1ffffffu) | 0x80000000u };
      u32x8 g1 = { (1u << 16) | (1u << 20) | (4u << 22) | (3u << 25),
                   (64u << 16),            // tensor_dim0[15:0] @ bits 63:48
                   (128u << 16),           // tensor_dim0 hi=0 | tensor_dim1 lo
                   (64u << 16),            // tensor_dim1 hi=0 | tile_dim0=64
                   128u,                   // tile_dim1=128 | tile_dim2=0
                   64u, 0u, 0u };          // tensor_dim0_stride=64
      asm volatile("tensor_load_to_lds %0, %1" :: "s"(g0), "s"(g1) : "memory");
    }

    // --- V tile: VMEM load + transposed LDS store ---
    for (int i = 0; i < 4; ++i) {
      int id = tid + i * 256;
      int r = id >> 3, c8 = (id & 7) * 8;
      uint4 vv = *(const uint4*)(Vh + ((size_t)(kc * CH + r)) * DH + c8);
      const __bf16* pv = (const __bf16*)&vv;
      for (int j = 0; j < 8; ++j) Vt[(c8 + j) * LDV + r] = pv[j];
    }

    // prefetch next chunk's K/V cachelines into L2 while we compute
    if (kc < qc) {
      const char* nk = (const char*)(Kh + (size_t)(kc + 1) * CH * DH);
      const char* nv = (const char*)(Vh + (size_t)(kc + 1) * CH * DH);
      __builtin_prefetch(nk + tid * 64, 0, 0);
      __builtin_prefetch(nv + tid * 64, 0, 0);
    }

    if (wave == 0) __builtin_amdgcn_s_wait_tensorcnt(0);
    __syncthreads();

    // --- S = Q @ K^T : 16 x 128 per wave, pipelined B-fragment loads ---
    f32x8 s[8];
    bf16x16 b0 = frag_b(Ks, LDK, lane);
    bf16x16 b1 = frag_b(Ks + 32, LDK, lane);
#pragma unroll
    for (int nt = 0; nt < 8; ++nt) {
      bf16x16 nb0 = b0, nb1 = b1;
      if (nt < 7) {
        nb0 = frag_b(Ks + (nt + 1) * 16 * LDK,      LDK, lane);
        nb1 = frag_b(Ks + (nt + 1) * 16 * LDK + 32, LDK, lane);
      }
      f32x8 c = {};
      c = wmma_bf16(qf0, b0, c);
      c = wmma_bf16(qf1, b1, c);
      s[nt] = c;
      b0 = nb0; b1 = nb1;
    }

    // scale + causal mask (diagonal chunk only)
    const bool diag = (kc == qc);
    for (int nt = 0; nt < 8; ++nt)
      for (int i = 0; i < 8; ++i) {
        float v = s[nt][i] * 0.125f;                       // 1/sqrt(64)
        if (diag) {
          int kpos = nt * 16 + ln;
          int qpos = wave * 16 + i + 8 * half;
          if (kpos > qpos) v = -__builtin_inff();
        }
        s[nt][i] = v;
      }

    // online softmax: row stats per (vgpr i, lane half), reduce over 16 lanes
    for (int i = 0; i < 8; ++i) {
      float rmax = s[0][i];
      for (int nt = 1; nt < 8; ++nt) rmax = fmaxf(rmax, s[nt][i]);
      for (int off = 1; off < 16; off <<= 1) rmax = fmaxf(rmax, __shfl_xor(rmax, off, 32));
      float mn = fmaxf(m[i], rmax);
      float alpha = __expf(m[i] - mn);                     // 0 on first chunk
      m[i] = mn;
      float rs = 0.f;
      for (int nt = 0; nt < 8; ++nt) {
        float p = __expf(s[nt][i] - mn);
        s[nt][i] = p;
        rs += p;
      }
      for (int off = 1; off < 16; off <<= 1) rs += __shfl_xor(rs, off, 32);
      l[i] = l[i] * alpha + rs;
      for (int ont = 0; ont < 4; ++ont) o[ont][i] = o[ont][i] * alpha;
    }

    __syncthreads();        // all waves finished reading Ks; safe to overlay with P
    __bf16* Pw = Ps + wave * 16 * LDP;
    for (int nt = 0; nt < 8; ++nt)
      for (int i = 0; i < 8; ++i)
        Pw[(i + 8 * half) * LDP + nt * 16 + ln] = (__bf16)s[nt][i];
    __syncthreads();        // C-layout -> A-layout transpose visible

    // --- O += P @ V : 4 k-steps x 4 n-tiles, pipelined V fragments ---
#pragma unroll
    for (int kt = 0; kt < 4; ++kt) {
      bf16x16 pf = frag_a(Pw + kt * 32, LDP, lane);
      bf16x16 vf = frag_b(Vt + kt * 32, LDV, lane);
#pragma unroll
      for (int ont = 0; ont < 4; ++ont) {
        bf16x16 nv = vf;
        if (ont < 3) nv = frag_b(Vt + (ont + 1) * 16 * LDV + kt * 32, LDV, lane);
        o[ont] = wmma_bf16(pf, vf, o[ont]);
        vf = nv;
      }
    }
  }

  // normalize and store merged (B, T, DM) bf16 for the output projection
  const int b = bh >> 4, h = bh & 15;
  for (int ont = 0; ont < 4; ++ont)
    for (int i = 0; i < 8; ++i) {
      int t = qc * CH + wave * 16 + i + 8 * half;
      int col = h * DH + ont * 16 + ln;
      float v = o[ont][i] / l[i];
      ((__bf16*)Og)[(((size_t)b * TT + t) << 10) + col] = (__bf16)v;
    }
}

// ---------------------------------------------------------------------------
extern "C" void kernel_launch(void* const* d_in, const int* in_sizes, int n_in,
                              void* d_out, int out_size, void* d_ws, size_t ws_size,
                              hipStream_t stream) {
  (void)in_sizes; (void)n_in; (void)out_size; (void)ws_size;
  const float* x  = (const float*)d_in[0];
  const float* Wq = (const float*)d_in[1];
  const float* Wk = (const float*)d_in[2];
  const float* Wv = (const float*)d_in[3];
  const float* Wo = (const float*)d_in[4];

  const size_t NEL = (size_t)MTOT * DM;                 // 16,777,216 elems
  unsigned short* Qb = (unsigned short*)d_ws;           // bf16 (B,H,T,dh)
  unsigned short* Kb = Qb + NEL;
  unsigned short* Vb = Kb + NEL;
  unsigned short* Ab = Vb + NEL;                        // bf16 merged (B,T,DM)

  dim3 gblk(256), ggrid(DM / 128, MTOT / 128);
  gemm_bf16<false, true><<<ggrid, gblk, 0, stream>>>((const void*)x, Wq, (void*)Qb);
  gemm_bf16<false, true><<<ggrid, gblk, 0, stream>>>((const void*)x, Wk, (void*)Kb);
  gemm_bf16<false, true><<<ggrid, gblk, 0, stream>>>((const void*)x, Wv, (void*)Vb);

  flash_kernel<<<dim3(TT / CH, BBATCH * NH), dim3(256), 0, stream>>>(Qb, Kb, Vb, Ab);

  gemm_bf16<true, false><<<ggrid, gblk, 0, stream>>>((const void*)Ab, Wo, d_out);
}
